// WindowAttention_63024350101939
// MI455X (gfx1250) — matrix operand order repaired and checked by reading, not compiled, not run on hardware
//
#include <hip/hip_runtime.h>
#include <hip/hip_bf16.h>
#include <math.h>
#include <stdint.h>

typedef __attribute__((ext_vector_type(16))) _Float16 v16h;
typedef __attribute__((ext_vector_type(8)))  _Float16 h8;
typedef __attribute__((ext_vector_type(8)))  float    v8f;
typedef __attribute__((ext_vector_type(8)))  float    f8;
typedef __attribute__((ext_vector_type(4)))  int      v4i;

#define EMBED 768
#define HEADS 12
#define DH    64
#define WIN   64
#define NW    64           // windows per batch = 4096/64
#define BATCH 8
#define NSEQ  4096
#define MROWS (BATCH*NSEQ) // 32768

#define AS1 __attribute__((address_space(1)))
#define AS3 __attribute__((address_space(3)))

#if defined(__has_builtin)
#if __has_builtin(__builtin_amdgcn_global_load_async_to_lds_b128)
#define HAVE_ASYNC_COPY 1
#endif
#if __has_builtin(__builtin_amdgcn_s_wait_asynccnt)
#define HAVE_WAIT_ASYNC 1
#endif
#endif

// Copy 16 bytes global -> LDS. Prefer the gfx1250 async-to-LDS path
// (ASYNCcnt-tracked, no VGPR data round trip); fall back to a plain
// register-staged copy if the builtin is unavailable.
__device__ __forceinline__ void lds_copy16B(_Float16* l, const _Float16* g) {
#ifdef HAVE_ASYNC_COPY
    __builtin_amdgcn_global_load_async_to_lds_b128(
        (AS1 v4i*)(uintptr_t)g,
        (AS3 v4i*)(uint32_t)(uintptr_t)l, 0, 0);
#else
    *(h8*)l = *(const h8*)g;
#endif
}

__device__ __forceinline__ void lds_copy_fence() {
#ifdef HAVE_ASYNC_COPY
#ifdef HAVE_WAIT_ASYNC
    __builtin_amdgcn_s_wait_asynccnt(0);
#else
    asm volatile("s_wait_asynccnt 0x0" ::: "memory");
#endif
#endif
}

__device__ __forceinline__ v8f wmma_f16(v16h a, v16h b, v8f c) {
    // D = A(16x32,f16) * B(32x16,f16) + C(16x16,f32)
    return __builtin_amdgcn_wmma_f32_16x16x32_f16(
        false, a, false, b, (short)0, c, false, false);
}

// A fragment (16x32 f16, row-major source, row stride ld):
// lane<16: row=lane,    K={0..7,16..23}; lane>=16: row=lane-16, K={8..15,24..31}
__device__ __forceinline__ v16h load_a_frag(const _Float16* base, int ld, int lane) {
    const int lr = lane & 15, hi = lane >> 4;
    const _Float16* p = base + lr * ld + hi * 8;
    v16h a;
    *((h8*)&a)       = *(const h8*)(p);
    *(((h8*)&a) + 1) = *(const h8*)(p + 16);
    return a;
}

// B fragment (32x16 f16). Row n of `base` holds column n of B (K contiguous).
// lane<16: col=lane, K={0..7,16..23}; lane>=16: col=lane-16, K={8..15,24..31}.
__device__ __forceinline__ v16h load_b_frag(const _Float16* base, int ld, int lane) {
    const int lr = lane & 15, hi = lane >> 4;
    const _Float16* p = base + lr * ld + hi * 8;
    v16h b;
    *((h8*)&b)       = *(const h8*)(p);
    *(((h8*)&b) + 1) = *(const h8*)(p + 16);
    return b;
}

// Vectorized f32 -> f16 conversion, 8 elements per thread.
__global__ __launch_bounds__(256) void cvt_f32_to_f16(
    const float* __restrict__ src, _Float16* __restrict__ dst, int n8) {
    int i = blockIdx.x * 256 + threadIdx.x;
    if (i < n8) {
        f8 v = ((const f8*)src)[i];
        h8 o;
#pragma unroll
        for (int k = 0; k < 8; ++k) o[k] = (_Float16)v[k];
        ((h8*)dst)[i] = o;
    }
}

// C = X[32768,768] * Wt[768,768]^T (+bias). Wt row-major [n,k] == K-major columns.
// EPI 0: f32 out to [row, col] (final projection)
// EPI 1: f16 out to head layout [b,w,h,win,dh]   (Q, K)
// EPI 2: f16 out to transposed  [b,w,h,dh,win]   (V)
// Double-buffered LDS staging: stage slab kk+1 (async) while computing slab kk.
template <int EPI>
__global__ __launch_bounds__(256) void gemm_proj(
    const _Float16* __restrict__ X, const _Float16* __restrict__ Wt,
    const float* __restrict__ bias, void* __restrict__ outp) {
    __shared__ _Float16 Xs[2][128 * 40]; // 128x32 tile, stride 40 (80B)
    __shared__ _Float16 Ws[2][64 * 40];  // 64x32 tile

    const int tid  = threadIdx.x;
    const int lane = tid & 31;
    const int wid  = tid >> 5;       // 8 waves
    const int m_base = blockIdx.y * 128;
    const int n_base = blockIdx.x * 64;
    const int lr = lane & 15, hi = lane >> 4;

    auto stage = [&](int buf, int k0) {
#pragma unroll
        for (int j = 0; j < 2; ++j) {
            int c = tid + j * 256;              // 512 chunks of 8 halves
            int row = c >> 2, kc = (c & 3) * 8;
            lds_copy16B(&Xs[buf][row * 40 + kc],
                        X + (size_t)(m_base + row) * EMBED + k0 + kc);
        }
        int row = tid >> 2, kc = (tid & 3) * 8; // 256 chunks
        lds_copy16B(&Ws[buf][row * 40 + kc],
                    Wt + (size_t)(n_base + row) * EMBED + k0 + kc);
    };

    v8f acc[4] = {};

    stage(0, 0);
    lds_copy_fence();
    __syncthreads();

    for (int kk = 0; kk < EMBED / 32; ++kk) {
        const int cur = kk & 1;
        if (kk + 1 < EMBED / 32) stage(cur ^ 1, (kk + 1) * 32);

        v16h a;
        {
            const _Float16* p = &Xs[cur][(wid * 16 + lr) * 40 + hi * 8];
            *((h8*)&a)       = *(const h8*)(p);
            *(((h8*)&a) + 1) = *(const h8*)(p + 16);
        }
#pragma unroll
        for (int t = 0; t < 4; ++t) {
            v16h b;
            const _Float16* p = &Ws[cur][(t * 16 + lr) * 40 + hi * 8];
            *((h8*)&b)       = *(const h8*)(p);
            *(((h8*)&b) + 1) = *(const h8*)(p + 16);
            acc[t] = wmma_f16(a, b, acc[t]);
        }
        lds_copy_fence();
        __syncthreads();
    }

    // Epilogue: C/D layout -> VGPR i, lane: row = 8*hi + i, col = lr (per tile)
    const int m_strip = m_base + wid * 16;
#pragma unroll
    for (int t = 0; t < 4; ++t) {
#pragma unroll
        for (int i = 0; i < 8; ++i) {
            const int row = m_strip + hi * 8 + i;
            const int col = n_base + t * 16 + lr;
            const float val = acc[t][i] + bias[col];
            if (EPI == 0) {
                ((float*)outp)[(size_t)row * EMBED + col] = val;
            } else {
                const int b = row >> 12, rem = row & 4095;
                const int w = rem >> 6, wm = rem & 63;
                const int h = col >> 6, d = col & 63;
                size_t off;
                if (EPI == 1)
                    off = (((size_t)((b * NW + w) * HEADS + h) * WIN + wm) * DH + d);
                else
                    off = (((size_t)((b * NW + w) * HEADS + h) * DH + d) * WIN + wm);
                ((_Float16*)outp)[off] = (_Float16)val;
            }
        }
    }
}

// One block per (b, w, h). 4 waves; wave owns a 16-row strip of the 64x64
// window. K and V tiles are staged once into LDS (all 4 waves reuse them).
__global__ __launch_bounds__(128) void win_attn(
    const _Float16* __restrict__ Qh, const _Float16* __restrict__ Kh,
    const _Float16* __restrict__ Vt, _Float16* __restrict__ Oh) {
    __shared__ _Float16 Kl[64 * 72];
    __shared__ _Float16 Vl[64 * 72];
    __shared__ _Float16 Pl[64 * 72];

    const int tid  = threadIdx.x;
    const int lane = tid & 31;
    const int wid  = tid >> 5;
    const int lr = lane & 15, hi = lane >> 4;

    const int bwh = blockIdx.x;
    const int h = bwh % HEADS;
    const int w = (bwh / HEADS) % NW;
    const int b = bwh / (HEADS * NW);
    const size_t hb = ((size_t)(b * NW + w) * HEADS + h) * (size_t)(WIN * DH);
    const _Float16* Q = Qh + hb;
    const _Float16* K = Kh + hb;
    const _Float16* V = Vt + hb; // [dh, win]: row d = column d of V

    // Stage K and V: each 64 rows x 64 halves = 512 chunks; 4 per thread.
#pragma unroll
    for (int j = 0; j < 4; ++j) {
        int c = tid + j * 128;
        int row = c >> 3, kc = (c & 7) * 8;
        lds_copy16B(&Kl[row * 72 + kc], K + row * DH + kc);
        lds_copy16B(&Vl[row * 72 + kc], V + row * DH + kc);
    }
    lds_copy_fence();
    __syncthreads();

    const int m0 = wid * 16;

    // S = Q Kt : 16x64 strip = 4 tiles x 2 K-steps
    v8f s[4] = {};
#pragma unroll
    for (int ks = 0; ks < 2; ++ks) {
        v16h a = load_a_frag(Q + m0 * DH + ks * 32, DH, lane);
#pragma unroll
        for (int t = 0; t < 4; ++t) {
            v16h bf = load_b_frag(Kl + (t * 16) * 72 + ks * 32, 72, lane);
            s[t] = wmma_f16(a, bf, s[t]);
        }
    }

    // Softmax over rows. Row (m0 + 8*hi + i) lives in s[t][i] across the 16
    // lanes sharing `hi`; xor masks 1..8 stay inside that half-group.
    const float scale = 0.125f; // 1/sqrt(64)
#pragma unroll
    for (int i = 0; i < 8; ++i) {
        float mx = -3.0e30f;
#pragma unroll
        for (int t = 0; t < 4; ++t) mx = fmaxf(mx, s[t][i]);
#pragma unroll
        for (int x = 1; x < 16; x <<= 1) mx = fmaxf(mx, __shfl_xor(mx, x, 32));
        float sum = 0.f;
#pragma unroll
        for (int t = 0; t < 4; ++t) {
            float e = __expf((s[t][i] - mx) * scale);
            s[t][i] = e;
            sum += e;
        }
#pragma unroll
        for (int x = 1; x < 16; x <<= 1) sum += __shfl_xor(sum, x, 32);
        const float inv = 1.0f / sum;
#pragma unroll
        for (int t = 0; t < 4; ++t)
            Pl[(m0 + hi * 8 + i) * 72 + t * 16 + lr] = (_Float16)(s[t][i] * inv);
    }
    __syncthreads();

    // O = P V : 4 d-tiles x 2 K-steps
#pragma unroll
    for (int td = 0; td < 4; ++td) {
        v8f o = {};
#pragma unroll
        for (int ks = 0; ks < 2; ++ks) {
            v16h a;
            const _Float16* p = Pl + (m0 + lr) * 72 + ks * 32 + hi * 8;
            *((h8*)&a)       = *(const h8*)(p);
            *(((h8*)&a) + 1) = *(const h8*)(p + 16);
            v16h bf = load_b_frag(Vl + (td * 16) * 72 + ks * 32, 72, lane);
            o = wmma_f16(a, bf, o);
        }
#pragma unroll
        for (int i = 0; i < 8; ++i) {
            const int row = b * NSEQ + w * WIN + m0 + hi * 8 + i;
            const int col = h * DH + td * 16 + lr;
            Oh[(size_t)row * EMBED + col] = (_Float16)o[i];
        }
    }
}

extern "C" void kernel_launch(void* const* d_in, const int* in_sizes, int n_in,
                              void* d_out, int out_size, void* d_ws, size_t ws_size,
                              hipStream_t stream) {
    const float* q  = (const float*)d_in[0];
    const float* Wq = (const float*)d_in[3];
    const float* bq = (const float*)d_in[4];
    const float* Wk = (const float*)d_in[5];
    const float* bk = (const float*)d_in[6];
    const float* Wv = (const float*)d_in[7];
    const float* bv = (const float*)d_in[8];
    const float* Wo = (const float*)d_in[9];
    const float* bo = (const float*)d_in[10];
    float* out = (float*)d_out;

    const size_t SZ  = (size_t)MROWS * EMBED;  // 25,165,824 halves per tensor
    const size_t WSZ = (size_t)EMBED * EMBED;  //    589,824 halves per weight

    _Float16* Xh  = (_Float16*)d_ws;
    _Float16* Qh  = Xh + SZ;
    _Float16* Kh  = Qh + SZ;
    _Float16* Vt  = Kh + SZ;
    _Float16* Wqh = Vt + SZ;
    _Float16* Wkh = Wqh + WSZ;
    _Float16* Wvh = Wkh + WSZ;
    _Float16* Woh = Wvh + WSZ;
    _Float16* Ah  = Xh; // attention output reuses Xh (X no longer needed then)

    const int n8x = (int)(SZ / 8), n8w = (int)(WSZ / 8);
    cvt_f32_to_f16<<<(n8x + 255) / 256, 256, 0, stream>>>(q, Xh, n8x);
    cvt_f32_to_f16<<<(n8w + 255) / 256, 256, 0, stream>>>(Wq, Wqh, n8w);
    cvt_f32_to_f16<<<(n8w + 255) / 256, 256, 0, stream>>>(Wk, Wkh, n8w);
    cvt_f32_to_f16<<<(n8w + 255) / 256, 256, 0, stream>>>(Wv, Wvh, n8w);
    cvt_f32_to_f16<<<(n8w + 255) / 256, 256, 0, stream>>>(Wo, Woh, n8w);

    dim3 gg(EMBED / 64, MROWS / 128); // (12, 256)
    gemm_proj<1><<<gg, 256, 0, stream>>>(Xh, Wqh, bq, (void*)Qh);
    gemm_proj<1><<<gg, 256, 0, stream>>>(Xh, Wkh, bk, (void*)Kh);
    gemm_proj<2><<<gg, 256, 0, stream>>>(Xh, Wvh, bv, (void*)Vt);

    win_attn<<<BATCH * NW * HEADS, 128, 0, stream>>>(Qh, Kh, Vt, Ah);

    gemm_proj<0><<<gg, 256, 0, stream>>>(Ah, Woh, bo, (void*)out);
}
